// TrajsEncoder2_39152921870520
// MI455X (gfx1250) — matrix operand order repaired and checked by reading, not compiled, not run on hardware
//
#include <hip/hip_runtime.h>

typedef __attribute__((ext_vector_type(16))) __bf16 v16bf;
typedef __attribute__((ext_vector_type(8)))  float  v8f;

// ---------- helpers ----------
__device__ __forceinline__ unsigned fenc(float f) {
  unsigned u = __float_as_uint(f);
  return (u & 0x80000000u) ? ~u : (u | 0x80000000u);
}
__device__ __forceinline__ float fdec(unsigned u) {
  return __uint_as_float((u & 0x80000000u) ? (u ^ 0x80000000u) : ~u);
}

// ---------- pack kernels: build WMMA B-operand blobs (bf16, ISA 32x16 lane layout) ----------
// layout per (chunk,ntile): 32 lanes x 16 bf16; lane l holds N=l&15, element s has K=(l>>4)*8 + s for s<8, 16+(l>>4)*8+(s-8) for s>=8
__global__ void k_pack_w2(const float* __restrict__ nnW2, const float* __restrict__ nnb2,
                          __bf16* __restrict__ blob) {
  int idx = blockIdx.x * 256 + threadIdx.x;        // over 2112*64
  if (idx >= 2112 * 64) return;
  int k = idx >> 6, n = idx & 63;
  float v;
  if (k < 2048) { int c = k >> 5, kk = k & 31; v = nnW2[kk * 4096 + c * 64 + n]; }
  else          { int i = k - 2048;            v = nnb2[i * 64 + n]; }
  int c = k >> 5, kk = k & 31, t = n >> 4, nn = n & 15;
  int lane = (((kk >> 3) & 1) << 4) | nn, j = ((kk >> 4) << 2) | ((kk & 7) >> 1), h = kk & 1;
  blob[(size_t)(c * 4 + t) * 512 + lane * 16 + j * 2 + h] = (__bf16)v;
}

__global__ void k_pack_gw1(const float* __restrict__ gW0, __bf16* __restrict__ blob) {
  int idx = blockIdx.x * 256 + threadIdx.x;        // over 192*256
  if (idx >= 192 * 256) return;
  int k = idx >> 8, n = idx & 255;
  float v = gW0[k * 256 + n];
  int c = k >> 5, kk = k & 31, t = n >> 4, nn = n & 15;
  int lane = (((kk >> 3) & 1) << 4) | nn, j = ((kk >> 4) << 2) | ((kk & 7) >> 1), h = kk & 1;
  blob[(size_t)(c * 16 + t) * 512 + lane * 16 + j * 2 + h] = (__bf16)v;
}

// ---------- K1: nodes_mlp (16->128 relu ->16) fused with @att_W -> xw[N,64] ----------
__global__ __launch_bounds__(256) void k_nodes(
    const float* __restrict__ x, const float* __restrict__ W0, const float* __restrict__ b0,
    const float* __restrict__ W1, const float* __restrict__ b1, const float* __restrict__ AW,
    float* __restrict__ xw, int N) {
  __shared__ float sW0[16 * 128], sb0[128], sW1[128 * 16], sb1[16], sAW[16 * 64];
  for (int i = threadIdx.x; i < 16 * 128; i += 256) sW0[i] = W0[i];
  for (int i = threadIdx.x; i < 128; i += 256)      sb0[i] = b0[i];
  for (int i = threadIdx.x; i < 128 * 16; i += 256) sW1[i] = W1[i];
  for (int i = threadIdx.x; i < 16; i += 256)       sb1[i] = b1[i];
  for (int i = threadIdx.x; i < 16 * 64; i += 256)  sAW[i] = AW[i];
  __syncthreads();
  int n = blockIdx.x * 256 + threadIdx.x;
  if (n >= N) return;
  float xi[16], ho[16];
#pragma unroll
  for (int i = 0; i < 16; ++i) xi[i] = x[n * 16 + i];
#pragma unroll
  for (int o = 0; o < 16; ++o) ho[o] = sb1[o];
  for (int j = 0; j < 128; ++j) {
    float t = sb0[j];
#pragma unroll
    for (int i = 0; i < 16; ++i) t += xi[i] * sW0[i * 128 + j];
    t = fmaxf(t, 0.f);
#pragma unroll
    for (int o = 0; o < 16; ++o) ho[o] += t * sW1[j * 16 + o];
  }
  for (int o = 0; o < 64; ++o) {
    float s = 0.f;
#pragma unroll
    for (int i = 0; i < 16; ++i) s += ho[i] * sAW[i * 64 + o];
    xw[(size_t)n * 64 + o] = s;
  }
}

// ---------- K2: edges_mlp (8->128 relu ->16) fused with nn layer1 (16->32 relu) -> nnh[E,32] ----------
__global__ __launch_bounds__(256) void k_edges(
    const float* __restrict__ ea, const float* __restrict__ W0, const float* __restrict__ b0,
    const float* __restrict__ W1, const float* __restrict__ b1,
    const float* __restrict__ nW0, const float* __restrict__ nb0,
    float* __restrict__ nnh, int E) {
  __shared__ float sW0[8 * 128], sb0[128], sW1[128 * 16], sb1[16], snW0[16 * 32], snb0[32];
  for (int i = threadIdx.x; i < 8 * 128; i += 256)  sW0[i] = W0[i];
  for (int i = threadIdx.x; i < 128; i += 256)      sb0[i] = b0[i];
  for (int i = threadIdx.x; i < 128 * 16; i += 256) sW1[i] = W1[i];
  for (int i = threadIdx.x; i < 16; i += 256)       sb1[i] = b1[i];
  for (int i = threadIdx.x; i < 16 * 32; i += 256)  snW0[i] = nW0[i];
  for (int i = threadIdx.x; i < 32; i += 256)       snb0[i] = nb0[i];
  __syncthreads();
  int e = blockIdx.x * 256 + threadIdx.x;
  if (e >= E) return;
  float xi[8], he[16];
#pragma unroll
  for (int i = 0; i < 8; ++i) xi[i] = ea[(size_t)e * 8 + i];
#pragma unroll
  for (int o = 0; o < 16; ++o) he[o] = sb1[o];
  for (int j = 0; j < 128; ++j) {
    float t = sb0[j];
#pragma unroll
    for (int i = 0; i < 8; ++i) t += xi[i] * sW0[i * 128 + j];
    t = fmaxf(t, 0.f);
#pragma unroll
    for (int o = 0; o < 16; ++o) he[o] += t * sW1[j * 16 + o];
  }
  for (int k = 0; k < 32; ++k) {
    float t = snb0[k];
#pragma unroll
    for (int i = 0; i < 16; ++i) t += he[i] * snW0[i * 32 + k];
    nnh[(size_t)e * 32 + k] = fmaxf(t, 0.f);
  }
}

// ---------- K3: GCN(sum) ----------
__global__ void k_count(const int* __restrict__ row, int* __restrict__ cnt, int E) {
  int e = blockIdx.x * 256 + threadIdx.x;
  if (e < E) atomicAdd(&cnt[row[e]], 1);
}
__global__ void k_gcn_init(const int* __restrict__ cnt, const float* __restrict__ xw,
                           const float* __restrict__ attb, float* __restrict__ x1,
                           float* __restrict__ dis, int N) {
  int tid = blockIdx.x * 256 + threadIdx.x;
  if (tid >= N * 64) return;
  int n = tid >> 6, o = tid & 63;
  float d = rsqrtf((float)cnt[n] + 2.f);
  if (o == 0) dis[n] = d;
  x1[tid] = 2.f * d * d * xw[tid] + attb[o];
}
__global__ void k_gcn_sum(const int* __restrict__ row, const int* __restrict__ col,
                          const float* __restrict__ dis, const float* __restrict__ xw,
                          float* __restrict__ x1, int E) {
  long long tid = (long long)blockIdx.x * 256 + threadIdx.x;
  if (tid >= (long long)E * 64) return;
  int e = (int)(tid >> 6), o = (int)(tid & 63);
  int r = row[e], c = col[e];
  float v = dis[r] * dis[c] * xw[(size_t)c * 64 + o];
  atomicAdd(&x1[(size_t)r * 64 + o], v);
}

// ---------- K4: fused NNConv  msg = P[E,2112] @ W2'[2112,64]  via bf16 WMMA ----------
// Software-pipelined: B tiles for chunk c+1 are in flight while chunk c's WMMAs issue.
#define K4W 4
__global__ __launch_bounds__(128) void k_nnconv(
    const float* __restrict__ x1, const float* __restrict__ nnh,
    const int* __restrict__ row, const int* __restrict__ col,
    const __bf16* __restrict__ blob, float* __restrict__ summsg, int E, int NT) {
  __shared__ float x1t[K4W][16 * 64];
  int wave = threadIdx.x >> 5, lane = threadIdx.x & 31;
  int tile = blockIdx.x * K4W + wave;
  if (tile >= NT) return;
  int e0 = tile * 16;
  float* xt = x1t[wave];
  for (int idx = lane; idx < 16 * 64; idx += 32) {       // stage x1[col[e]] (wave-private)
    int el = idx >> 6, o = idx & 63;
    int e = e0 + el;
    float v = 0.f;
    if (e < E) v = x1[(size_t)col[e] * 64 + o];
    xt[idx] = v;
  }
  int m = lane & 15, kb = (lane >> 4) * 8, hh = lane >> 4, nn = lane & 15;
  int e = e0 + m;
  float nv[16];
#pragma unroll
  for (int j = 0; j < 8; ++j) {
    nv[j]     = (e < E) ? nnh[(size_t)e * 32 + kb + j]      : 0.f;
    nv[8 + j] = (e < E) ? nnh[(size_t)e * 32 + 16 + kb + j] : 0.f;
  }
  const __bf16* bp = blob + (size_t)lane * 16;
  v8f acc[4];
#pragma unroll
  for (int t = 0; t < 4; ++t)
#pragma unroll
    for (int i = 0; i < 8; ++i) acc[t][i] = 0.f;

  v16bf bA[4], bB[4];
#pragma unroll
  for (int t = 0; t < 4; ++t) bA[t] = *(const v16bf*)(bp + (size_t)t * 512);

  for (int c = 0; c < 64; c += 2) {
    // prefetch chunk c+1
#pragma unroll
    for (int t = 0; t < 4; ++t) bB[t] = *(const v16bf*)(bp + (size_t)((c + 1) * 4 + t) * 512);
    {
      float s = xt[m * 64 + c];
      v16bf a;
#pragma unroll
      for (int idx = 0; idx < 16; ++idx) a[idx] = (__bf16)(s * nv[idx]);
#pragma unroll
      for (int t = 0; t < 4; ++t)
        acc[t] = __builtin_amdgcn_wmma_f32_16x16x32_bf16(false, a, false, bA[t], (short)0,
                                                         acc[t], false, false);
    }
    // prefetch chunk c+2 (c+2 <= 64, chunk 64 exists)
#pragma unroll
    for (int t = 0; t < 4; ++t) bA[t] = *(const v16bf*)(bp + (size_t)((c + 2) * 4 + t) * 512);
    {
      float s = xt[m * 64 + c + 1];
      v16bf a;
#pragma unroll
      for (int idx = 0; idx < 16; ++idx) a[idx] = (__bf16)(s * nv[idx]);
#pragma unroll
      for (int t = 0; t < 4; ++t)
        acc[t] = __builtin_amdgcn_wmma_f32_16x16x32_bf16(false, a, false, bB[t], (short)0,
                                                         acc[t], false, false);
    }
  }
  // bias chunks 64,65: P[m,kk] = x1c[m, (c-64)*32+kk]; bA holds chunk 64 already
#pragma unroll
  for (int t = 0; t < 4; ++t) bB[t] = *(const v16bf*)(bp + (size_t)(65 * 4 + t) * 512);
#pragma unroll
  for (int cb = 0; cb < 2; ++cb) {
    int base = cb * 32;
    v16bf a;
#pragma unroll
    for (int idx = 0; idx < 16; ++idx) {
      int kk = (idx < 8) ? (kb + idx) : (16 + kb + idx - 8);
      a[idx] = (__bf16)xt[m * 64 + base + kk];
    }
#pragma unroll
    for (int t = 0; t < 4; ++t)
      acc[t] = __builtin_amdgcn_wmma_f32_16x16x32_bf16(false, a, false, (cb ? bB[t] : bA[t]),
                                                       (short)0, acc[t], false, false);
  }
#pragma unroll
  for (int j = 0; j < 8; ++j) {
    int me = e0 + j + 8 * hh;                        // D row M = j + 8*(lane>>4)
    if (me < E) {
      int r = row[me];
#pragma unroll
      for (int t = 0; t < 4; ++t)
        atomicAdd(&summsg[(size_t)r * 64 + t * 16 + nn], acc[t][j]);
    }
  }
}

// ---------- K5: x2 = tanh(mean + x1 @ nn_root + nn_b) ----------
__global__ __launch_bounds__(256) void k_x2(
    const float* __restrict__ summsg, const int* __restrict__ cnt, const float* __restrict__ x1,
    const float* __restrict__ R, const float* __restrict__ rb, float* __restrict__ x2, int N) {
  __shared__ float sR[64 * 64], sb[64];
  for (int i = threadIdx.x; i < 4096; i += 256) sR[i] = R[i];
  if (threadIdx.x < 64) sb[threadIdx.x] = rb[threadIdx.x];
  __syncthreads();
  int tid = blockIdx.x * 256 + threadIdx.x;
  if (tid >= N * 64) return;
  int n = tid >> 6, o = tid & 63;
  float mean = summsg[tid] / fmaxf((float)cnt[n], 1.f);
  float s = sb[o];
  const float* xr = x1 + (size_t)n * 64;
  for (int i = 0; i < 64; ++i) s += xr[i] * sR[i * 64 + o];
  x2[tid] = tanhf(mean + s);
}

// ---------- K6: GCN(max) ----------
__global__ __launch_bounds__(256) void k_gcn2_init(
    const float* __restrict__ x2, const float* __restrict__ W, const float* __restrict__ dis,
    float* __restrict__ xw2, unsigned* __restrict__ enc3, int N) {
  __shared__ float sW[64 * 64];
  for (int i = threadIdx.x; i < 4096; i += 256) sW[i] = W[i];
  __syncthreads();
  int tid = blockIdx.x * 256 + threadIdx.x;
  if (tid >= N * 64) return;
  int n = tid >> 6, o = tid & 63;
  float s = 0.f;
  const float* xr = x2 + (size_t)n * 64;
  for (int i = 0; i < 64; ++i) s += xr[i] * sW[i * 64 + o];
  xw2[tid] = s;
  float d = dis[n];
  enc3[tid] = fenc(2.f * d * d * s);
}
__global__ void k_gcn_max(const int* __restrict__ row, const int* __restrict__ col,
                          const float* __restrict__ dis, const float* __restrict__ xw2,
                          unsigned* __restrict__ enc3, int E) {
  long long tid = (long long)blockIdx.x * 256 + threadIdx.x;
  if (tid >= (long long)E * 64) return;
  int e = (int)(tid >> 6), o = (int)(tid & 63);
  int r = row[e], c = col[e];
  float v = dis[r] * dis[c] * xw2[(size_t)c * 64 + o];
  atomicMax(&enc3[(size_t)r * 64 + o], fenc(v));
}
__global__ void k_xc(const float* __restrict__ x1, const float* __restrict__ x2,
                     const unsigned* __restrict__ enc3, const float* __restrict__ lastb,
                     float* __restrict__ xc, int N) {
  long long tid = (long long)blockIdx.x * 256 + threadIdx.x;
  if (tid >= (long long)N * 192) return;
  int n = (int)(tid / 192), o = (int)(tid % 192);
  float v;
  if (o < 64)       v = x1[(size_t)n * 64 + o];
  else if (o < 128) v = x2[(size_t)n * 64 + (o - 64)];
  else              v = fdec(enc3[(size_t)n * 64 + (o - 128)]) + lastb[o - 128];
  xc[tid] = v;
}

// ---------- K7: gate MLP layer1 [N,192]@[192,256] via WMMA, fused relu + layer2 dot ----------
#define K7W 4
__global__ __launch_bounds__(128) void k_gate(
    const float* __restrict__ xc, const __bf16* __restrict__ blob,
    const float* __restrict__ gb1, const float* __restrict__ gw2,
    const float* __restrict__ gb2, float* __restrict__ gate, int N, int NT) {
  __shared__ float xct[K7W][16 * 192];
  int wave = threadIdx.x >> 5, lane = threadIdx.x & 31;
  int tile = blockIdx.x * K7W + wave;
  if (tile >= NT) return;
  int n0 = tile * 16;
  float* xt = xct[wave];
  for (int idx = lane; idx < 16 * 192; idx += 32) {
    int nl = idx / 192, o = idx % 192;
    int n = n0 + nl;
    xt[idx] = (n < N) ? xc[(size_t)n * 192 + o] : 0.f;
  }
  int m = lane & 15, kb = (lane >> 4) * 8, hh = lane >> 4, nn = lane & 15;
  const __bf16* bp = blob + (size_t)lane * 16;
  v8f acc[16];
#pragma unroll
  for (int t = 0; t < 16; ++t)
#pragma unroll
    for (int i = 0; i < 8; ++i) acc[t][i] = 0.f;
  for (int c = 0; c < 6; ++c) {
    v16bf a;
#pragma unroll
    for (int idx = 0; idx < 16; ++idx) {
      int kk = (idx < 8) ? (kb + idx) : (16 + kb + idx - 8);
      a[idx] = (__bf16)xt[m * 192 + c * 32 + kk];
    }
    // two groups of 8 B tiles: loads batched ahead of their WMMAs
    v16bf bb[8];
#pragma unroll
    for (int t = 0; t < 8; ++t) bb[t] = *(const v16bf*)(bp + (size_t)((c * 16 + t) * 32) * 16);
#pragma unroll
    for (int t = 0; t < 8; ++t)
      acc[t] = __builtin_amdgcn_wmma_f32_16x16x32_bf16(false, a, false, bb[t], (short)0,
                                                       acc[t], false, false);
#pragma unroll
    for (int t = 0; t < 8; ++t) bb[t] = *(const v16bf*)(bp + (size_t)((c * 16 + 8 + t) * 32) * 16);
#pragma unroll
    for (int t = 0; t < 8; ++t)
      acc[8 + t] = __builtin_amdgcn_wmma_f32_16x16x32_bf16(false, a, false, bb[t], (short)0,
                                                           acc[8 + t], false, false);
  }
  float p[8];
#pragma unroll
  for (int j = 0; j < 8; ++j) p[j] = 0.f;
#pragma unroll
  for (int t = 0; t < 16; ++t) {
    int o = t * 16 + nn;
    float bo = gb1[o], wo = gw2[o];
#pragma unroll
    for (int j = 0; j < 8; ++j) {
      float h = fmaxf(acc[t][j] + bo, 0.f);
      p[j] += h * wo;
    }
  }
#pragma unroll
  for (int j = 0; j < 8; ++j) {
    float v = p[j];
    for (int off = 1; off < 16; off <<= 1) v += __shfl_xor(v, off, 32);
    p[j] = v;
  }
  if (nn == 0) {
    float b2v = gb2[0];
#pragma unroll
    for (int j = 0; j < 8; ++j) {
      int n = n0 + j + 8 * hh;
      if (n < N) gate[n] = p[j] + b2v;
    }
  }
}

// ---------- K8: global-attention pooling (batch is sorted, NPG nodes per graph) ----------
__global__ __launch_bounds__(128) void k_pool(
    const float* __restrict__ xc, const float* __restrict__ gate,
    float* __restrict__ pooled, int NPG) {
  int b = blockIdx.x, t = threadIdx.x;
  int n0 = b * NPG;
  __shared__ float red[128];
  float mx = -3.4e38f;
  for (int i = t; i < NPG; i += 128) mx = fmaxf(mx, gate[n0 + i]);
  red[t] = mx; __syncthreads();
  for (int s = 64; s > 0; s >>= 1) { if (t < s) red[t] = fmaxf(red[t], red[t + s]); __syncthreads(); }
  float gm = red[0]; __syncthreads();
  float sw = 0.f;
  for (int i = t; i < NPG; i += 128) sw += expf(gate[n0 + i] - gm);
  red[t] = sw; __syncthreads();
  for (int s = 64; s > 0; s >>= 1) { if (t < s) red[t] += red[t + s]; __syncthreads(); }
  float ws = red[0];
  for (int o = t; o < 192; o += 128) {
    float s = 0.f;
    for (int i = 0; i < NPG; ++i)
      s += expf(gate[n0 + i] - gm) * xc[(size_t)(n0 + i) * 192 + o];
    pooled[(size_t)b * 192 + o] = s / ws;
  }
}

// ---------- K9: out MLP [196 -> 32 relu -> 16 relu -> 16] ----------
__global__ __launch_bounds__(32) void k_out(
    const float* __restrict__ pooled, const float* __restrict__ scales,
    const float* __restrict__ orient,
    const float* __restrict__ W0, const float* __restrict__ b0,
    const float* __restrict__ W1, const float* __restrict__ b1,
    const float* __restrict__ W2, const float* __restrict__ b2,
    float* __restrict__ out) {
  int b = blockIdx.x, t = threadIdx.x;
  __shared__ float feat[196], h1[32], h2[16];
  for (int i = t; i < 192; i += 32) feat[i] = pooled[(size_t)b * 192 + i];
  if (t == 0) feat[192] = logf(scales[b] + 1e-5f);
  if (t < 3)  feat[193 + t] = orient[b * 3 + t];
  __syncthreads();
  { float s = b0[t]; for (int i = 0; i < 196; ++i) s += feat[i] * W0[i * 32 + t]; h1[t] = fmaxf(s, 0.f); }
  __syncthreads();
  if (t < 16) { float s = b1[t]; for (int i = 0; i < 32; ++i) s += h1[i] * W1[i * 16 + t]; h2[t] = fmaxf(s, 0.f); }
  __syncthreads();
  if (t < 16) { float s = b2[t]; for (int i = 0; i < 16; ++i) s += h2[i] * W2[i * 16 + t]; out[(size_t)b * 16 + t] = s; }
}

// ---------- launch ----------
extern "C" void kernel_launch(void* const* d_in, const int* in_sizes, int n_in,
                              void* d_out, int out_size, void* d_ws, size_t ws_size,
                              hipStream_t stream) {
  const float* x      = (const float*)d_in[0];
  const float* eattr  = (const float*)d_in[1];
  const int*   row    = (const int*)d_in[2];
  const int*   col    = (const int*)d_in[3];
  const float* scales = (const float*)d_in[5];
  const float* orient = (const float*)d_in[6];
  const float *nW0 = (const float*)d_in[7],  *nb0 = (const float*)d_in[8];
  const float *nW1 = (const float*)d_in[9],  *nb1 = (const float*)d_in[10];
  const float *eW0 = (const float*)d_in[11], *eb0 = (const float*)d_in[12];
  const float *eW1 = (const float*)d_in[13], *eb1 = (const float*)d_in[14];
  const float *attW = (const float*)d_in[15], *attb = (const float*)d_in[16];
  const float *mW0 = (const float*)d_in[17], *mb0 = (const float*)d_in[18];
  const float *mW1 = (const float*)d_in[19], *mb1 = (const float*)d_in[20];
  const float *nnR = (const float*)d_in[21], *nnBb = (const float*)d_in[22];
  const float *lW  = (const float*)d_in[23], *lb  = (const float*)d_in[24];
  const float *gW0 = (const float*)d_in[25], *gb0 = (const float*)d_in[26];
  const float *gW1 = (const float*)d_in[27], *gb1s = (const float*)d_in[28];
  const float *oW0 = (const float*)d_in[29], *ob0 = (const float*)d_in[30];
  const float *oW1 = (const float*)d_in[31], *ob1 = (const float*)d_in[32];
  const float *oW2 = (const float*)d_in[33], *ob2 = (const float*)d_in[34];

  const int N = in_sizes[0] / 16;
  const int E = in_sizes[2];
  const int B = in_sizes[5];       // NS == 1
  const int NPG = N / B;

  char* ws = (char*)d_ws;
  size_t off = 0;
  auto alloc = [&](size_t bytes) -> char* {
    char* p = ws + off;
    off = (off + bytes + 255) & ~(size_t)255;
    return p;
  };
  float*  xw      = (float*)alloc((size_t)N * 64 * 4);   // reused as xw2
  float*  x1      = (float*)alloc((size_t)N * 64 * 4);
  float*  nnh     = (float*)alloc((size_t)E * 32 * 4);
  float*  summsg  = (float*)alloc((size_t)N * 64 * 4);   // reused as enc3 (unsigned)
  float*  x2      = (float*)alloc((size_t)N * 64 * 4);
  float*  xc      = (float*)alloc((size_t)N * 192 * 4);
  int*    cnt     = (int*)  alloc((size_t)N * 4);
  float*  dis     = (float*)alloc((size_t)N * 4);
  float*  gate    = (float*)alloc((size_t)N * 4);
  float*  pooled  = (float*)alloc((size_t)B * 192 * 4);
  __bf16* w2blob  = (__bf16*)alloc((size_t)2112 * 64 * 2);
  __bf16* gw1blob = (__bf16*)alloc((size_t)192 * 256 * 2);
  (void)ws_size; (void)n_in; (void)out_size;

  hipMemsetAsync(cnt, 0, (size_t)N * 4, stream);
  hipMemsetAsync(summsg, 0, (size_t)N * 64 * 4, stream);

  k_pack_w2 <<<(2112 * 64 + 255) / 256, 256, 0, stream>>>(mW1, mb1, w2blob);
  k_pack_gw1<<<(192 * 256 + 255) / 256, 256, 0, stream>>>(gW0, gw1blob);

  k_nodes<<<(N + 255) / 256, 256, 0, stream>>>(x, nW0, nb0, nW1, nb1, attW, xw, N);
  k_edges<<<(E + 255) / 256, 256, 0, stream>>>(eattr, eW0, eb0, eW1, eb1, mW0, mb0, nnh, E);

  k_count<<<(E + 255) / 256, 256, 0, stream>>>(row, cnt, E);
  k_gcn_init<<<((size_t)N * 64 + 255) / 256, 256, 0, stream>>>(cnt, xw, attb, x1, dis, N);
  k_gcn_sum<<<(unsigned)(((long long)E * 64 + 255) / 256), 256, 0, stream>>>(row, col, dis, xw, x1, E);

  const int NT4 = (E + 15) / 16;
  k_nnconv<<<(NT4 + K4W - 1) / K4W, 32 * K4W, 0, stream>>>(x1, nnh, row, col, w2blob, summsg, E, NT4);

  k_x2<<<((size_t)N * 64 + 255) / 256, 256, 0, stream>>>(summsg, cnt, x1, nnR, nnBb, x2, N);

  k_gcn2_init<<<((size_t)N * 64 + 255) / 256, 256, 0, stream>>>(x2, lW, dis, xw, (unsigned*)summsg, N);
  k_gcn_max<<<(unsigned)(((long long)E * 64 + 255) / 256), 256, 0, stream>>>(row, col, dis, xw, (unsigned*)summsg, E);
  k_xc<<<(unsigned)(((long long)N * 192 + 255) / 256), 256, 0, stream>>>(x1, x2, (const unsigned*)summsg, lb, xc, N);

  const int NT7 = (N + 15) / 16;
  k_gate<<<(NT7 + K7W - 1) / K7W, 32 * K7W, 0, stream>>>(xc, gw1blob, gb0, gW1, gb1s, gate, N, NT7);

  k_pool<<<B, 128, 0, stream>>>(xc, gate, pooled, NPG);
  k_out<<<B, 32, 0, stream>>>(pooled, scales, orient, oW0, ob0, oW1, ob1, oW2, ob2, (float*)d_out);
}
